// FLNO1d_30004641530289
// MI455X (gfx1250) — matrix operand order repaired and checked by reading, not compiled, not run on hardware
//
#include <hip/hip_runtime.h>
#include <math.h>

// FLNO-1d on gfx1250 (MI455X, wave32, WMMA).
// All three per-layer branches are linear in h -> fused into one frequency-
// domain transfer function G[i,o,k]; per-layer pipeline:
//   FFT(h) -> per-k complex GEMM via v_wmma_f32_16x16x32_f16 -> IFFT -> +bias -> GELU
// A-fragment packs Re(alpha) in rows 0-7 and Im(alpha) in rows 8-15 so one
// pair of WMMAs per 16-wide output tile produces all four real products;
// re/im are then combined with a single v_permlanex16 cross-half swap and
// stored branchlessly (lower lanes -> re, upper lanes -> im).
// G is pre-swizzled into WMMA B-fragment layout -> per-lane b128 loads.

#define LSEQ   2048
#define WCH    32
#define BATCH  8
#define MODES  16
#define NLAYER 4

typedef __attribute__((ext_vector_type(16))) _Float16 v16h;
typedef __attribute__((ext_vector_type(8)))  float    v8f;

__device__ __forceinline__ float gelu_exact(float v) {
  return 0.5f * v * (1.0f + erff(v * 0.70710678118654752f));
}

// Cross-half swap (lane ^ 16) as pure VALU: v_permlanex16_b32 with identity
// nibble selects (no LDS round-trip, no DScnt).
__device__ __forceinline__ float swap_half16(float v) {
  int r = __builtin_amdgcn_permlanex16(__float_as_int(v), __float_as_int(v),
                                       0x76543210, 0xfedcba98, false, false);
  return __int_as_float(r);
}

// A-fragment (16x32 f16) element -> K index map: lane holds row M=lane&15;
// element e -> K = ((lane>>4)<<4) + e  (contiguous, matches B fragments)
__device__ __forceinline__ int a_frag_k(int e, int lanehalf) {
  return (lanehalf << 4) + e;
}

// ---------------------------------------------------------------- fc0 + grid
__global__ __launch_bounds__(256)
void fc0_kernel(const float* __restrict__ x, const float* __restrict__ fc0_w,
                const float* __restrict__ fc0_b, float* __restrict__ h) {
  int idx = blockIdx.x * blockDim.x + threadIdx.x;           // over B*W*L
  if (idx >= BATCH * WCH * LSEQ) return;
  int l = idx % LSEQ;
  int w = (idx / LSEQ) % WCH;
  int b = idx / (LSEQ * WCH);
  float xi = x[b * LSEQ + l];
  float gi = (float)l * (1.0f / (float)(LSEQ - 1));          // linspace(0,1,L)
  float v  = fc0_w[w * 2 + 0] * xi + fc0_w[w * 2 + 1] * gi + fc0_b[w];
  h[(b * WCH + w) * LSEQ + l] = gelu_exact(v);               // (B,W,L)
}

// ------------------------------------------- fused transfer function G
// G = Hw (pole-residue) + conv_w[o,i] + spectral weights (Hermitian-mirrored),
// written straight into WMMA B-fragment layout:
//   GF[((k*2 + (o>>4)) * 32 + lane) * 16 + e],  lane = (o&15) + 16*(i>=16),
//   e = i&15   (B is 32x16: row K=i, col N=o within each 16-wide o-tile)
__global__ __launch_bounds__(256)
void build_G_kernel(const float* __restrict__ t,
                    const float* __restrict__ pole_re, const float* __restrict__ pole_im,
                    const float* __restrict__ res_re,  const float* __restrict__ res_im,
                    const float* __restrict__ spec_re, const float* __restrict__ spec_im,
                    const float* __restrict__ conv_w,  int layer,
                    _Float16* __restrict__ GreF, _Float16* __restrict__ GimF) {
  int idx = blockIdx.x * blockDim.x + threadIdx.x;           // k*W*W + i*W + o
  if (idx >= LSEQ * WCH * WCH) return;
  int o = idx % WCH;
  int i = (idx / WCH) % WCH;
  int k = idx / (WCH * WCH);

  float dt = t[1] - t[0];
  float fk = (k < LSEQ / 2) ? (float)k : (float)(k - LSEQ);  // fftfreq * L
  float w  = 6.28318530717958647f * (fk * (1.0f / (float)LSEQ)) / dt;  // lam = i*w

  const int base = ((layer * WCH + i) * WCH + o) * MODES;
  float hr = 0.0f, hi = 0.0f;
  #pragma unroll
  for (int m = 0; m < MODES; m++) {
    float rr = res_re[base + m], ri = res_im[base + m];
    float c  = -pole_re[base + m];                            // Re(lam - pole)
    float d  = w - pole_im[base + m];                         // Im(lam - pole)
    float inv = 1.0f / (c * c + d * d);
    hr += (rr * c + ri * d) * inv;
    hi += (ri * c - rr * d) * inv;
  }
  hr += conv_w[(layer * WCH + o) * WCH + i];                  // 1x1 conv, merged

  if (k < MODES) {                                            // spectral, k>=0 half
    hr += spec_re[base + k];
    hi += spec_im[base + k];
  } else if (k > LSEQ - MODES) {                              // Hermitian mirror
    int kk = LSEQ - k;
    hr += spec_re[base + kk];
    hi -= spec_im[base + kk];
  }
  int lane = (o & 15) + 16 * (i >= 16);
  int foff = ((k * 2 + (o >> 4)) * 32 + lane) * 16 + (i & 15);
  GreF[foff] = (_Float16)hr;
  GimF[foff] = (_Float16)hi;
}

// ----------------------------------------------- 2048-pt Stockham FFT in LDS
// Autosorting radix-2; ping-pong buffers; 11 stages -> result lands in buf1.
__device__ float2* stockham2048(float2* __restrict__ buf0, float2* __restrict__ buf1,
                                float sign) {
  float2* src = buf0;
  float2* dst = buf1;
  int s = 1, ls = 0;
  for (int ncur = LSEQ; ncur > 1; ncur >>= 1) {
    int   half   = ncur >> 1;
    float theta0 = sign * 6.28318530717958647f / (float)ncur;
    for (int tl = threadIdx.x; tl < LSEQ / 2; tl += blockDim.x) {
      int p = tl >> ls;
      int q = tl & (s - 1);
      float sn, cs;
      __sincosf(theta0 * (float)p, &sn, &cs);
      float2 u = src[q + (p << ls)];
      float2 v = src[q + ((p + half) << ls)];
      float2 amb = make_float2(u.x - v.x, u.y - v.y);
      dst[q + ((2 * p    ) << ls)] = make_float2(u.x + v.x, u.y + v.y);
      dst[q + ((2 * p + 1) << ls)] = make_float2(amb.x * cs - amb.y * sn,
                                                 amb.x * sn + amb.y * cs);
    }
    __syncthreads();
    float2* tmp = src; src = dst; dst = tmp;
    s <<= 1; ls++;
  }
  return src;
}

__global__ __launch_bounds__(256)
void fft_fwd_kernel(const float* __restrict__ h, float2* __restrict__ alphaC) {
  __shared__ float2 buf0[LSEQ];
  __shared__ float2 buf1[LSEQ];
  int row = blockIdx.x;                                      // b*W + i
  for (int j = threadIdx.x; j < LSEQ; j += blockDim.x)
    buf0[j] = make_float2(h[row * LSEQ + j], 0.0f);
  __syncthreads();
  float2* r = stockham2048(buf0, buf1, -1.0f);               // forward: e^{-i..}
  for (int j = threadIdx.x; j < LSEQ; j += blockDim.x)
    alphaC[row * LSEQ + j] = r[j];                           // interleaved b64
}

__global__ __launch_bounds__(256)
void ifft_gelu_kernel(const float2* __restrict__ gammaC,
                      const float* __restrict__ conv_b, int layer,
                      float* __restrict__ hout) {
  __shared__ float2 buf0[LSEQ];
  __shared__ float2 buf1[LSEQ];
  int row = blockIdx.x;                                      // b*W + o
  int o   = row % WCH;
  for (int j = threadIdx.x; j < LSEQ; j += blockDim.x)
    buf0[j] = gammaC[row * LSEQ + j];
  __syncthreads();
  float2* r = stockham2048(buf0, buf1, +1.0f);               // inverse: e^{+i..}
  float bias = conv_b[layer * WCH + o];
  for (int j = threadIdx.x; j < LSEQ; j += blockDim.x)
    hout[row * LSEQ + j] = gelu_exact(r[j].x * (1.0f / (float)LSEQ) + bias);
}

// -------------------------------- per-frequency complex GEMM via WMMA (f16)
// gamma[b,o,k] = sum_i alpha[b,i,k] * G[i,o,k];  one wave per frequency k.
// A packs Re(alpha) in M=0..7, Im(alpha) in M=8..15:
//   C1 = A*Gr -> {ar*Gr | ai*Gr},  C2 = A*Gi -> {ar*Gi | ai*Gi}
// For row r: lower half has x1=ar*Gr, x2=ar*Gi; upper half x1=ai*Gr, x2=ai*Gi.
// One cross-half swap of x2 gives:  lower: gr = x1 - swap(x2)
//                                   upper: gi = x1 + swap(x2)
// -> branchless: all 32 lanes store one b32 (lower=re, upper=im).
__global__ __launch_bounds__(128)
void freq_gemm_kernel(const float2* __restrict__ alphaC,
                      const _Float16* __restrict__ GreF, const _Float16* __restrict__ GimF,
                      float2* __restrict__ gammaC) {
  int wave = threadIdx.x >> 5;
  int lane = threadIdx.x & 31;
  int k    = blockIdx.x * 4 + wave;

  // warm L2 for the next block's fragments (global_prefetch_b8)
  if (k + 4 < LSEQ) {
    __builtin_prefetch((const void*)(GreF + (size_t)(k + 4) * 1024), 0, 1);
    __builtin_prefetch((const void*)(GimF + (size_t)(k + 4) * 1024), 0, 1);
  }

  // ---- packed A fragment (scalar b32 gathers; comp folds into the address)
  int M        = lane & 15;
  int lanehalf = lane >> 4;
  int b        = M & 7;                                      // batch row
  int comp     = M >> 3;                                     // 0 = Re, 1 = Im
  const float* ap = (const float*)alphaC;
  v16h af;
  #pragma unroll
  for (int e = 0; e < 16; e++) {
    int K = a_frag_k(e, lanehalf);                           // channel i
    af[e] = (_Float16)ap[((b * WCH + K) * LSEQ + k) * 2 + comp];
  }

  float* gf = (float*)gammaC;
  #pragma unroll
  for (int nt = 0; nt < 2; nt++) {                           // o in two 16-tiles
    // B fragments pre-swizzled: 16 contiguous halves per lane (2x b128)
    const v16h br = *(const v16h*)(GreF + (((size_t)k * 2 + nt) * 32 + lane) * 16);
    const v16h bi = *(const v16h*)(GimF + (((size_t)k * 2 + nt) * 32 + lane) * 16);

    v8f c1 = {}, c2 = {};
    c1 = __builtin_amdgcn_wmma_f32_16x16x32_f16(false, af, false, br, (short)0, c1, false, false);
    c2 = __builtin_amdgcn_wmma_f32_16x16x32_f16(false, af, false, bi, (short)0, c2, false, false);

    int o0 = nt * 16;
    #pragma unroll
    for (int r = 0; r < 8; r++) {                            // output row b=r
      float x1 = c1[r];
      float y2 = swap_half16(c2[r]);                         // v_permlanex16
      float val = lanehalf ? (x1 + y2) : (x1 - y2);          // upper: gi, lower: gr
      gf[((r * WCH + o0 + M) * LSEQ + k) * 2 + lanehalf] = val;
    }
  }
}

// ---------------------------- fc1 (WMMA, K=32) + GELU + fc2 (dot) fused head
__global__ __launch_bounds__(256)
void fc12_kernel(const float* __restrict__ h, const float* __restrict__ fc1_w,
                 const float* __restrict__ fc1_b, const float* __restrict__ fc2_w,
                 const float* __restrict__ fc2_b, float* __restrict__ out) {
  __shared__ float acc[16];
  int lane = threadIdx.x & 31;
  int wave = threadIdx.x >> 5;                               // 8 waves -> 8 n-tiles
  int r0   = blockIdx.x * 16;                                // rows g = b*L + l
  if (threadIdx.x < 16) acc[threadIdx.x] = 0.0f;
  __syncthreads();

  int M = lane & 15, lanehalf = lane >> 4;
  int g = r0 + M, b = g / LSEQ, l = g % LSEQ;
  v16h af;
  #pragma unroll
  for (int e = 0; e < 16; e++) {
    int K = a_frag_k(e, lanehalf);                           // channel w
    af[e] = (_Float16)h[(b * WCH + K) * LSEQ + l];
  }
  int n0 = wave * 16, N = lane & 15, kh = lane >> 4;
  v16h bf;
  #pragma unroll
  for (int e = 0; e < 16; e++) {
    int K = (kh << 4) + e;
    bf[e] = (_Float16)fc1_w[(n0 + N) * WCH + K];             // fc1_w[n][w]
  }
  v8f c = {};
  c = __builtin_amdgcn_wmma_f32_16x16x32_f16(false, af, false, bf, (short)0, c, false, false);

  #pragma unroll
  for (int r = 0; r < 8; r++) {
    int Mr = r + ((lane >= 16) ? 8 : 0);
    int n  = n0 + (lane & 15);
    float p = gelu_exact(c[r] + fc1_b[n]) * fc2_w[n];
    #pragma unroll
    for (int s = 8; s >= 1; s >>= 1) p += __shfl_xor(p, s, 16);  // half-wave reduce
    if ((lane & 15) == 0) atomicAdd(&acc[Mr], p);                // ds_add_f32
  }
  __syncthreads();
  if (threadIdx.x < 16) out[r0 + threadIdx.x] = acc[threadIdx.x] + fc2_b[0];
}

// --------------------------------------------------------------------- launch
extern "C" void kernel_launch(void* const* d_in, const int* in_sizes, int n_in,
                              void* d_out, int out_size, void* d_ws, size_t ws_size,
                              hipStream_t stream) {
  const float* x       = (const float*)d_in[0];
  const float* t       = (const float*)d_in[1];
  const float* fc0_w   = (const float*)d_in[2];
  const float* fc0_b   = (const float*)d_in[3];
  const float* pole_re = (const float*)d_in[4];
  const float* pole_im = (const float*)d_in[5];
  const float* res_re  = (const float*)d_in[6];
  const float* res_im  = (const float*)d_in[7];
  const float* spec_re = (const float*)d_in[8];
  const float* spec_im = (const float*)d_in[9];
  const float* conv_w  = (const float*)d_in[10];
  const float* conv_b  = (const float*)d_in[11];
  const float* fc1_w   = (const float*)d_in[12];
  const float* fc1_b   = (const float*)d_in[13];
  const float* fc2_w   = (const float*)d_in[14];
  const float* fc2_b   = (const float*)d_in[15];

  char* ws = (char*)d_ws;
  const size_t HN = (size_t)BATCH * WCH * LSEQ;              // 524288 elements
  float*    hA     = (float*)(ws);                           // 2 MB
  float*    hB     = (float*)(ws + HN * 4);                  // 2 MB
  float2*   alphaC = (float2*)(ws + HN * 8);                 // 4 MB
  float2*   gammaC = (float2*)(ws + HN * 16);                // 4 MB
  _Float16* GreF   = (_Float16*)(ws + HN * 24);              // 4 MB (frag layout)
  _Float16* GimF   = (_Float16*)(ws + HN * 28);              // 4 MB -> 20 MB total

  fc0_kernel<<<(int)(HN / 256), 256, 0, stream>>>(x, fc0_w, fc0_b, hA);

  float* hc = hA;
  float* hn = hB;
  for (int layer = 0; layer < NLAYER; layer++) {
    build_G_kernel<<<(LSEQ * WCH * WCH) / 256, 256, 0, stream>>>(
        t, pole_re, pole_im, res_re, res_im, spec_re, spec_im, conv_w, layer, GreF, GimF);
    fft_fwd_kernel<<<BATCH * WCH, 256, 0, stream>>>(hc, alphaC);
    freq_gemm_kernel<<<LSEQ / 4, 128, 0, stream>>>(alphaC, GreF, GimF, gammaC);
    ifft_gelu_kernel<<<BATCH * WCH, 256, 0, stream>>>(gammaC, conv_b, layer, hn);
    float* tmp = hc; hc = hn; hn = tmp;
  }

  fc12_kernel<<<(BATCH * LSEQ) / 16, 256, 0, stream>>>(
      hc, fc1_w, fc1_b, fc2_w, fc2_b, (float*)d_out);
}
